// FaceRecon_86457691668771
// MI455X (gfx1250) — matrix operand order repaired and checked by reading, not compiled, not run on hardware
//
#include <hip/hip_runtime.h>
#include <math.h>

// ---------------------------------------------------------------------------
// Problem constants (from reference)
// ---------------------------------------------------------------------------
#define BSZ   4
#define NV    2048
#define NV1   512
#define NV2   128
#define SUP   7
#define NBRK  10
#define OBJC  6
#define FACEC 30
#define BN_EPS 1e-5f

typedef __attribute__((ext_vector_type(16))) __bf16 bf16x16;
typedef __attribute__((ext_vector_type(8)))  __bf16 bf16x8;
typedef __attribute__((ext_vector_type(8)))  float  f32x8;

// ---------------------------------------------------------------------------
// Staging: fp32 -> bf16 with K padded to 32 (activations, row-major M x Kp)
// ---------------------------------------------------------------------------
__global__ void stage_act_k(const float* __restrict__ A, __bf16* __restrict__ Ab,
                            int M, int K, int Kp) {
    size_t i = (size_t)blockIdx.x * blockDim.x + threadIdx.x;
    size_t total = (size_t)M * Kp;
    if (i >= total) return;
    int k = (int)(i % Kp);
    size_t m = i / Kp;
    Ab[i] = (k < K) ? (__bf16)A[m * (size_t)K + k] : (__bf16)0.0f;
}

// Weights: fp32 (K x N) -> bf16 transposed+padded (Np x Kp), zero padded
__global__ void stage_weight_k(const float* __restrict__ W, __bf16* __restrict__ Wt,
                               int K, int N, int Kp, int Np) {
    size_t i = (size_t)blockIdx.x * blockDim.x + threadIdx.x;
    size_t total = (size_t)Np * Kp;
    if (i >= total) return;
    int k = (int)(i % Kp);
    int n = (int)(i / Kp);
    Wt[i] = (k < K && n < N) ? (__bf16)W[(size_t)k * N + n] : (__bf16)0.0f;
}

// ---------------------------------------------------------------------------
// WMMA GEMM: C(MxN,f32) = A(MxKp,bf16) * B^T(NpxKp,bf16) + bias
// One wave -> 16x64 C strip (4 accumulators), reusing each A fragment for
// 4 WMMAs.  4 waves/block cover 256 adjacent columns.  Np padded to 64.
// Fragment K order: contiguous k = 16*(lane/16) + s in both A and B (16-bit A 16x32; B 32x16; f32 C 16x16).
// ---------------------------------------------------------------------------
__global__ __launch_bounds__(128) void wmma_gemm_bf16(
    const __bf16* __restrict__ A,   // M x Kp
    const __bf16* __restrict__ Bt,  // Np x Kp  (transposed weights)
    const float* __restrict__ bias, // N
    float* __restrict__ C,          // M x N
    int M, int N, int Kp, int Np)
{
    int nbase = (blockIdx.x * 4 + threadIdx.y) * 64;
    if (nbase >= Np) return;
    int m0 = blockIdx.y * 16;
    int lane = threadIdx.x;
    int rl = lane & 15;     // row (A) or column (B/C) within tile
    int hf = lane >> 4;     // lane-half selects K sub-range

    const __bf16* arow = A + (size_t)(m0 + rl) * Kp + hf * 16;
    const __bf16* brow0 = Bt + (size_t)(nbase + rl) * Kp + hf * 16;
    const __bf16* brow1 = brow0 + (size_t)16 * Kp;
    const __bf16* brow2 = brow0 + (size_t)32 * Kp;
    const __bf16* brow3 = brow0 + (size_t)48 * Kp;

    f32x8 acc0 = {}, acc1 = {}, acc2 = {}, acc3 = {};
    for (int kb = 0; kb < Kp; kb += 32) {
        bf16x8 a0 = *(const bf16x8*)(arow + kb);        // K = kb + hf*16 + (0..7)
        bf16x8 a1 = *(const bf16x8*)(arow + kb + 8);    // K = kb + hf*16 + 8 + (0..7)
        bf16x16 av;
#pragma unroll
        for (int i = 0; i < 8; ++i) { av[i] = a0[i]; av[i + 8] = a1[i]; }
        bf16x16 b0 = *(const bf16x16*)(brow0 + kb);     // K = kb + hf*16 + (0..15)
        bf16x16 b1 = *(const bf16x16*)(brow1 + kb);
        bf16x16 b2 = *(const bf16x16*)(brow2 + kb);
        bf16x16 b3 = *(const bf16x16*)(brow3 + kb);
        acc0 = __builtin_amdgcn_wmma_f32_16x16x32_bf16(false, av, false, b0, (short)0, acc0, false, false);
        acc1 = __builtin_amdgcn_wmma_f32_16x16x32_bf16(false, av, false, b1, (short)0, acc1, false, false);
        acc2 = __builtin_amdgcn_wmma_f32_16x16x32_bf16(false, av, false, b2, (short)0, acc2, false, false);
        acc3 = __builtin_amdgcn_wmma_f32_16x16x32_bf16(false, av, false, b3, (short)0, acc3, false, false);
    }

#pragma unroll
    for (int j = 0; j < 4; ++j) {
        int col = nbase + j * 16 + rl;
        if (col < N) {
            f32x8 acc = (j == 0) ? acc0 : (j == 1) ? acc1 : (j == 2) ? acc2 : acc3;
            float bv = bias ? bias[col] : 0.0f;
#pragma unroll
            for (int r = 0; r < 8; ++r) {
                int row = m0 + r + hf * 8;
                C[(size_t)row * N + col] = acc[r] + bv;
            }
        }
    }
}

// ---------------------------------------------------------------------------
// Normalize columns of a 3 x C direction matrix
// ---------------------------------------------------------------------------
__global__ void norm_cols3_k(const float* __restrict__ d, float* __restrict__ o, int C) {
    int c = blockIdx.x * blockDim.x + threadIdx.x;
    if (c >= C) return;
    float x = d[c], y = d[C + c], z = d[2 * C + c];
    float n = fmaxf(sqrtf(x * x + y * y + z * z), 1e-12f);
    o[c] = x / n; o[C + c] = y / n; o[2 * C + c] = z / n;
}

// ---------------------------------------------------------------------------
// Brute-force KNN: per (b,v) keep k+1 smallest squared distances (drop self).
// Strict '<' insertion preserves lowest-index tie-break like top_k.
// ---------------------------------------------------------------------------
__global__ void knn_k(const float* __restrict__ verts, int* __restrict__ out,
                      int B, int Vn, int k) {
    int gid = blockIdx.x * blockDim.x + threadIdx.x;
    if (gid >= B * Vn) return;
    int b = gid / Vn, v = gid - b * Vn;
    const float* vb = verts + (size_t)b * Vn * 3;
    float px = vb[v * 3], py = vb[v * 3 + 1], pz = vb[v * 3 + 2];
    float bd[NBRK + 1]; int bi[NBRK + 1];
    int kk = k + 1;
#pragma unroll
    for (int j = 0; j < NBRK + 1; ++j) { bd[j] = 3.4e38f; bi[j] = 0; }
    for (int w = 0; w < Vn; ++w) {
        float dx = vb[w * 3] - px, dy = vb[w * 3 + 1] - py, dz = vb[w * 3 + 2] - pz;
        float d = dx * dx + dy * dy + dz * dz;
        if (d < bd[kk - 1]) {
            int j = kk - 1;
            while (j > 0 && d < bd[j - 1]) { bd[j] = bd[j - 1]; bi[j] = bi[j - 1]; --j; }
            bd[j] = d; bi[j] = w;
        }
    }
    for (int j = 0; j < k; ++j) out[(size_t)gid * k + j] = bi[j + 1];
}

// ---------------------------------------------------------------------------
// conv_surface: out[b,v,c] = relu( sum_s max_n relu(dir(b,v,n) . dn[:,s*outc+c]) )
// block = outc threads, grid = B*Vn
// ---------------------------------------------------------------------------
__global__ void conv_surface_k(const float* __restrict__ verts, const int* __restrict__ idx,
                               const float* __restrict__ dn, float* __restrict__ out,
                               int B, int Vn, int nn, int outc) {
    int bv = blockIdx.x;
    int b = bv / Vn, v = bv - b * Vn;
    __shared__ float sd[NBRK * 3];
    int t = threadIdx.x;
    const float* vb = verts + (size_t)b * Vn * 3;
    if (t < nn) {
        int w = idx[(size_t)bv * nn + t];
        float dx = vb[w * 3] - vb[v * 3];
        float dy = vb[w * 3 + 1] - vb[v * 3 + 1];
        float dz = vb[w * 3 + 2] - vb[v * 3 + 2];
        float n = fmaxf(sqrtf(dx * dx + dy * dy + dz * dz), 1e-12f);
        sd[t * 3] = dx / n; sd[t * 3 + 1] = dy / n; sd[t * 3 + 2] = dz / n;
    }
    __syncthreads();
    int C = SUP * outc;
    float acc = 0.f;
    for (int s = 0; s < SUP; ++s) {
        int col = s * outc + t;
        float cx = dn[col], cy = dn[C + col], cz = dn[2 * C + col];
        float m = 0.f;  // relu then max: min possible is 0
        for (int q = 0; q < nn; ++q)
            m = fmaxf(m, sd[q * 3] * cx + sd[q * 3 + 1] * cy + sd[q * 3 + 2] * cz);
        acc += m;
    }
    out[(size_t)bv * outc + t] = fmaxf(acc, 0.f);  // fused outer relu
}

// ---------------------------------------------------------------------------
// conv_layer combine: fc + sum_s max_n( relu(theta) * fs[nbr, s*outc+c] )
// fo: (B,Vn,8*outc) with fc = [0:outc], fs = [outc:8*outc]
// ---------------------------------------------------------------------------
__global__ void conv_combine_k(const float* __restrict__ verts, const int* __restrict__ idx,
                               const float* __restrict__ dn, const float* __restrict__ fo,
                               float* __restrict__ out, int B, int Vn, int nn, int outc) {
    int bv = blockIdx.x;
    int b = bv / Vn, v = bv - b * Vn;
    __shared__ float sd[NBRK * 3];
    __shared__ int si[NBRK];
    int t = threadIdx.x;
    const float* vb = verts + (size_t)b * Vn * 3;
    if (t < nn) {
        int w = idx[(size_t)bv * nn + t];
        si[t] = w;
        float dx = vb[w * 3] - vb[v * 3];
        float dy = vb[w * 3 + 1] - vb[v * 3 + 1];
        float dz = vb[w * 3 + 2] - vb[v * 3 + 2];
        float n = fmaxf(sqrtf(dx * dx + dy * dy + dz * dz), 1e-12f);
        sd[t * 3] = dx / n; sd[t * 3 + 1] = dy / n; sd[t * 3 + 2] = dz / n;
    }
    __syncthreads();
    int C = SUP * outc;
    int fst = 8 * outc;
    const float* fob = fo + (size_t)b * Vn * fst;
    float acc = fob[(size_t)v * fst + t];  // fc
    for (int s = 0; s < SUP; ++s) {
        int col = s * outc + t;
        float cx = dn[col], cy = dn[C + col], cz = dn[2 * C + col];
        float m = -3.4e38f;
        for (int q = 0; q < nn; ++q) {
            float th = fmaxf(sd[q * 3] * cx + sd[q * 3 + 1] * cy + sd[q * 3 + 2] * cz, 0.f);
            float fsv = fob[(size_t)si[q] * fst + outc + col];
            m = fmaxf(m, th * fsv);
        }
        acc += m;
    }
    out[(size_t)bv * outc + t] = acc;
}

// ---------------------------------------------------------------------------
// BatchNorm over rows (axes 0,1): one block per channel
// ---------------------------------------------------------------------------
__global__ void bn_stats_k(const float* __restrict__ x, float* __restrict__ mv,
                           int rows, int C) {
    int c = blockIdx.x;
    __shared__ float s1[256], s2[256];
    float a = 0.f, q = 0.f;
    for (int r = threadIdx.x; r < rows; r += blockDim.x) {
        float v = x[(size_t)r * C + c];
        a += v; q += v * v;
    }
    s1[threadIdx.x] = a; s2[threadIdx.x] = q;
    __syncthreads();
    for (int st = 128; st > 0; st >>= 1) {
        if ((int)threadIdx.x < st) { s1[threadIdx.x] += s1[threadIdx.x + st]; s2[threadIdx.x] += s2[threadIdx.x + st]; }
        __syncthreads();
    }
    if (threadIdx.x == 0) {
        float m = s1[0] / rows;
        mv[c] = m;
        mv[C + c] = fmaxf(s2[0] / rows - m * m, 0.f);
    }
}

__global__ void bn_apply_k(float* __restrict__ x, const float* __restrict__ mv,
                           const float* __restrict__ g, const float* __restrict__ be,
                           size_t total, int C, int relu) {
    size_t i = (size_t)blockIdx.x * blockDim.x + threadIdx.x;
    if (i >= total) return;
    int c = (int)(i % C);
    float val = g[c] * (x[i] - mv[c]) * rsqrtf(mv[C + c] + BN_EPS) + be[c];
    x[i] = relu ? fmaxf(val, 0.f) : val;
}

// ---------------------------------------------------------------------------
// Max-pool over 4 KNN neighbors
// ---------------------------------------------------------------------------
__global__ void pool_max4_k(const float* __restrict__ fm, const int* __restrict__ pidx,
                            float* __restrict__ out, int B, int Vn, int C) {
    size_t i = (size_t)blockIdx.x * blockDim.x + threadIdx.x;
    size_t total = (size_t)B * Vn * C;
    if (i >= total) return;
    int c = (int)(i % C);
    size_t bv = i / C;
    int b = (int)(bv / Vn);
    const float* fb = fm + (size_t)b * Vn * C;
    const int* pb = pidx + bv * 4;
    float m = -3.4e38f;
#pragma unroll
    for (int j = 0; j < 4; ++j) m = fmaxf(m, fb[(size_t)pb[j] * C + c]);
    out[i] = m;
}

// dst[b,i,c] = src[b, perm[i], c]
__global__ void gather_perm_k(const float* __restrict__ src, const int* __restrict__ perm,
                              float* __restrict__ dst, int B, int Vs, int Vd, int C) {
    size_t i = (size_t)blockIdx.x * blockDim.x + threadIdx.x;
    size_t total = (size_t)B * Vd * C;
    if (i >= total) return;
    int c = (int)(i % C);
    size_t bv = i / C;
    int b = (int)(bv / Vd);
    int vd = (int)(bv - (size_t)b * Vd);
    dst[i] = src[((size_t)b * Vs + perm[vd]) * C + c];
}

// per (b,vt) argmin distance to source set
__global__ void nearest_idx_k(const float* __restrict__ tgt, const float* __restrict__ src,
                              int* __restrict__ out, int B, int Vt, int Vs) {
    int gid = blockIdx.x * blockDim.x + threadIdx.x;
    if (gid >= B * Vt) return;
    int b = gid / Vt, v = gid - b * Vt;
    const float* tb = tgt + (size_t)b * Vt * 3;
    const float* sb = src + (size_t)b * Vs * 3;
    float px = tb[v * 3], py = tb[v * 3 + 1], pz = tb[v * 3 + 2];
    float best = 3.4e38f; int bi = 0;
    for (int w = 0; w < Vs; ++w) {
        float dx = sb[w * 3] - px, dy = sb[w * 3 + 1] - py, dz = sb[w * 3 + 2] - pz;
        float d = dx * dx + dy * dy + dz * dz;
        if (d < best) { best = d; bi = w; }
    }
    out[gid] = bi;
}

// dst[b,v,c] = src[b, nidx[b*Vt+v], c]
__global__ void gather_bv_k(const float* __restrict__ src, const int* __restrict__ nidx,
                            float* __restrict__ dst, int B, int Vt, int Vs, int C) {
    size_t i = (size_t)blockIdx.x * blockDim.x + threadIdx.x;
    size_t total = (size_t)B * Vt * C;
    if (i >= total) return;
    int c = (int)(i % C);
    size_t bv = i / C;
    int b = (int)(bv / Vt);
    dst[i] = src[((size_t)b * Vs + nidx[bv]) * C + c];
}

// copy (rows x C) block into strided destination at channel offset
__global__ void copy_seg_k(const float* __restrict__ src, float* __restrict__ dst,
                           size_t rows, int C, int stride, int doff) {
    size_t i = (size_t)blockIdx.x * blockDim.x + threadIdx.x;
    size_t total = rows * C;
    if (i >= total) return;
    size_t r = i / C; int c = (int)(i % C);
    dst[r * stride + doff + c] = src[i];
}

__global__ void onehot_k(const int* __restrict__ cat, float* __restrict__ feat,
                         int B, int Vn) {
    size_t i = (size_t)blockIdx.x * blockDim.x + threadIdx.x;
    size_t total = (size_t)B * Vn * OBJC;
    if (i >= total) return;
    int j = (int)(i % OBJC);
    size_t bv = i / OBJC;
    int b = (int)(bv / Vn);
    feat[bv * (1280 + OBJC) + 1280 + j] = (j == cat[b]) ? 1.0f : 0.0f;
}

// ff[b,c] = mean_v feat[b,v,c]  (c < 1280; feat row stride 1286)
__global__ void mean_feat_k(const float* __restrict__ feat, float* __restrict__ ff,
                            int B, int Vn) {
    int i = blockIdx.x * blockDim.x + threadIdx.x;
    if (i >= B * 1280) return;
    int b = i / 1280, c = i - b * 1280;
    const float* fb = feat + (size_t)b * Vn * 1286;
    float a = 0.f;
    for (int v = 0; v < Vn; ++v) a += fb[(size_t)v * 1286 + c];
    ff[i] = a / (float)Vn;
}

// y[b,v,:] = concat(ff[b,:1280], verts[b,v,:3])
__global__ void build_y_k(const float* __restrict__ ff, const float* __restrict__ verts,
                          float* __restrict__ y, int B, int Vn) {
    size_t i = (size_t)blockIdx.x * blockDim.x + threadIdx.x;
    size_t total = (size_t)B * Vn * 1283;
    if (i >= total) return;
    int c = (int)(i % 1283);
    size_t bv = i / 1283;
    int b = (int)(bv / Vn);
    y[i] = (c < 1280) ? ff[(size_t)b * 1280 + c]
                      : verts[bv * 3 + (c - 1280)];
}

// ---------------------------------------------------------------------------
// Host-side helpers
// ---------------------------------------------------------------------------
static inline void stage_w(hipStream_t s, const float* W, __bf16* Wt, int K, int N) {
    int Kp = (K + 31) & ~31, Np = (N + 63) & ~63;
    size_t total = (size_t)Np * Kp;
    stage_weight_k<<<(unsigned)((total + 255) / 256), 256, 0, s>>>(W, Wt, K, N, Kp, Np);
}

static inline void run_gemm(hipStream_t s, const float* A, int M, int K,
                            const __bf16* Wt, int N, const float* bias,
                            float* C, __bf16* Abf) {
    int Kp = (K + 31) & ~31, Np = (N + 63) & ~63;
    size_t total = (size_t)M * Kp;
    stage_act_k<<<(unsigned)((total + 255) / 256), 256, 0, s>>>(A, Abf, M, K, Kp);
    dim3 grid((Np / 64 + 3) / 4, M / 16);
    wmma_gemm_bf16<<<grid, dim3(32, 4), 0, s>>>(Abf, Wt, bias, C, M, N, Kp, Np);
}

static inline void run_bn(hipStream_t s, float* x, int rows, int C,
                          const float* g, const float* be, float* mv, int relu) {
    bn_stats_k<<<C, 256, 0, s>>>(x, mv, rows, C);
    size_t total = (size_t)rows * C;
    bn_apply_k<<<(unsigned)((total + 255) / 256), 256, 0, s>>>(x, mv, g, be, total, C, relu);
}

extern "C" void kernel_launch(void* const* d_in, const int* in_sizes, int n_in,
                              void* d_out, int out_size, void* d_ws, size_t ws_size,
                              hipStream_t stream) {
    (void)in_sizes; (void)n_in; (void)out_size; (void)ws_size;
    // ---- inputs (setup_inputs dict order; params flattened in insertion order)
    const float* verts = (const float*)d_in[0];
    const float* p_d0  = (const float*)d_in[1];
    const float* p_w1  = (const float*)d_in[2];
    const float* p_b1  = (const float*)d_in[3];
    const float* p_d1  = (const float*)d_in[4];
    const float* p_g1  = (const float*)d_in[5];
    const float* p_be1 = (const float*)d_in[6];
    const float* p_w2  = (const float*)d_in[7];
    const float* p_b2  = (const float*)d_in[8];
    const float* p_d2  = (const float*)d_in[9];
    const float* p_g2  = (const float*)d_in[10];
    const float* p_be2 = (const float*)d_in[11];
    const float* p_w3  = (const float*)d_in[12];
    const float* p_b3  = (const float*)d_in[13];
    const float* p_d3  = (const float*)d_in[14];
    const float* p_g3  = (const float*)d_in[15];
    const float* p_be3 = (const float*)d_in[16];
    const float* p_w4  = (const float*)d_in[17];
    const float* p_b4  = (const float*)d_in[18];
    const float* p_d4  = (const float*)d_in[19];
    const float* p_cw1 = (const float*)d_in[20];
    const float* p_cb1 = (const float*)d_in[21];
    const float* p_cg1 = (const float*)d_in[22];
    const float* p_ce1 = (const float*)d_in[23];
    const float* p_cw2 = (const float*)d_in[24];
    const float* p_cb2 = (const float*)d_in[25];
    const float* p_cg2 = (const float*)d_in[26];
    const float* p_ce2 = (const float*)d_in[27];
    const float* p_cw3 = (const float*)d_in[28];
    const float* p_cb3 = (const float*)d_in[29];
    const float* p_cg3 = (const float*)d_in[30];
    const float* p_ce3 = (const float*)d_in[31];
    const float* p_rw1 = (const float*)d_in[32];
    const float* p_rb1 = (const float*)d_in[33];
    const float* p_rg1 = (const float*)d_in[34];
    const float* p_re1 = (const float*)d_in[35];
    const float* p_rw2 = (const float*)d_in[36];
    const float* p_rb2 = (const float*)d_in[37];
    const float* p_fw1 = (const float*)d_in[38];
    const float* p_fb1 = (const float*)d_in[39];
    const float* p_fg1 = (const float*)d_in[40];
    const float* p_fe1 = (const float*)d_in[41];
    const float* p_fw2 = (const float*)d_in[42];
    const float* p_fb2 = (const float*)d_in[43];
    const float* p_fg2 = (const float*)d_in[44];
    const float* p_fe2 = (const float*)d_in[45];
    const float* p_fw3 = (const float*)d_in[46];
    const float* p_fb3 = (const float*)d_in[47];
    const float* p_fg3 = (const float*)d_in[48];
    const float* p_fe3 = (const float*)d_in[49];
    const float* p_fw4 = (const float*)d_in[50];
    const float* p_fb4 = (const float*)d_in[51];
    const int* cat_id  = (const int*)d_in[52];
    const int* perm1   = (const int*)d_in[53];
    const int* perm2   = (const int*)d_in[54];

    float* out = (float*)d_out;
    float* recon = out;                                   // (4,2048,3)
    float* face  = out + (size_t)BSZ * NV * 3;            // (4,2048,30)
    float* feat  = face + (size_t)BSZ * NV * FACEC;       // (4,2048,1286)

    // ---- workspace arena (256B aligned carve)
    char* ws = (char*)d_ws;
    size_t off = 0;
    auto alloc = [&](size_t bytes) -> void* {
        void* p = ws + off;
        off = (off + bytes + 255) & ~(size_t)255;
        return p;
    };
    const size_t M0 = (size_t)BSZ * NV;    // 8192
    const size_t M1 = (size_t)BSZ * NV1;   // 2048
    const size_t M2 = (size_t)BSZ * NV2;   // 512

    float* d0n = (float*)alloc(3 * SUP * 128 * 4);
    float* d1n = (float*)alloc(3 * SUP * 128 * 4);
    float* d2n = (float*)alloc(3 * SUP * 256 * 4);
    float* d3n = (float*)alloc(3 * SUP * 256 * 4);
    float* d4n = (float*)alloc(3 * SUP * 512 * 4);
    float* fm0 = (float*)alloc(M0 * 128 * 4);
    float* fm1 = (float*)alloc(M0 * 128 * 4);
    float* slab1 = (float*)alloc(M0 * 1283 * 4);  // fo1 (M0x1024) -> later y (M0x1283)
    float* slab2 = (float*)alloc(M0 * 512 * 4);   // fo2 (M1x2048) -> later x1 (M0x512)
    float* slab3 = (float*)alloc(M0 * 512 * 4);   // fo3 (M1x2048) -> later x2
    float* slab4 = (float*)alloc(M0 * 256 * 4);   // fo4 (M2x4096) -> later x3
    float* pooledF = (float*)alloc(M0 * 128 * 4); // pooled full -> later r1
    float* v1b = (float*)alloc(M1 * 3 * 4);
    float* f1b = (float*)alloc(M1 * 128 * 4);
    float* fm2 = (float*)alloc(M1 * 256 * 4);
    float* fm3 = (float*)alloc(M1 * 256 * 4);
    float* pooled1 = (float*)alloc(M1 * 256 * 4);
    float* v2b = (float*)alloc(M2 * 3 * 4);
    float* f2b = (float*)alloc(M2 * 256 * 4);
    float* fm4 = (float*)alloc(M2 * 512 * 4);
    float* fm2u = (float*)alloc(M0 * 256 * 4);    // later y2
    float* fm3u = (float*)alloc(M0 * 256 * 4);    // later y3
    float* fm4u = (float*)alloc(M0 * 512 * 4);    // later y1
    float* ffb  = (float*)alloc((size_t)BSZ * 1280 * 4);
    float* mv   = (float*)alloc(2 * 512 * 4);
    int* idx0  = (int*)alloc(M0 * NBRK * 4);
    int* pidx0 = (int*)alloc(M0 * 4 * 4);
    int* idx1  = (int*)alloc(M1 * NBRK * 4);
    int* pidx1 = (int*)alloc(M1 * 4 * 4);
    int* idx2  = (int*)alloc(M2 * NBRK * 4);
    int* nidx1 = (int*)alloc(M0 * 4);
    int* nidx2 = (int*)alloc(M0 * 4);
    // bf16 weight arena (transposed, N padded to 64, K padded to 32)
    __bf16* wt_w1  = (__bf16*)alloc((size_t)1024 * 128 * 2);
    __bf16* wt_w2  = (__bf16*)alloc((size_t)2048 * 128 * 2);
    __bf16* wt_w3  = (__bf16*)alloc((size_t)2048 * 256 * 2);
    __bf16* wt_w4  = (__bf16*)alloc((size_t)4096 * 256 * 2);
    __bf16* wt_cw1 = (__bf16*)alloc((size_t)512 * 1312 * 2);
    __bf16* wt_cw2 = (__bf16*)alloc((size_t)512 * 512 * 2);
    __bf16* wt_cw3 = (__bf16*)alloc((size_t)256 * 512 * 2);
    __bf16* wt_rw1 = (__bf16*)alloc((size_t)128 * 256 * 2);
    __bf16* wt_rw2 = (__bf16*)alloc((size_t)64 * 128 * 2);
    __bf16* wt_fw1 = (__bf16*)alloc((size_t)512 * 1312 * 2);
    __bf16* wt_fw2 = (__bf16*)alloc((size_t)256 * 512 * 2);
    __bf16* wt_fw3 = (__bf16*)alloc((size_t)128 * 256 * 2);
    __bf16* wt_fw4 = (__bf16*)alloc((size_t)64 * 128 * 2);
    __bf16* Abf = (__bf16*)alloc(M0 * 1312 * 2);  // activation staging (max M*Kp)

    hipStream_t s = stream;

    // ---- stage weights to bf16 (transposed + padded)
    stage_w(s, p_w1, wt_w1, 128, 1024);
    stage_w(s, p_w2, wt_w2, 128, 2048);
    stage_w(s, p_w3, wt_w3, 256, 2048);
    stage_w(s, p_w4, wt_w4, 256, 4096);
    stage_w(s, p_cw1, wt_cw1, 1286, 512);
    stage_w(s, p_cw2, wt_cw2, 512, 512);
    stage_w(s, p_cw3, wt_cw3, 512, 256);
    stage_w(s, p_rw1, wt_rw1, 256, 128);
    stage_w(s, p_rw2, wt_rw2, 128, 3);
    stage_w(s, p_fw1, wt_fw1, 1283, 512);
    stage_w(s, p_fw2, wt_fw2, 512, 256);
    stage_w(s, p_fw3, wt_fw3, 256, 128);
    stage_w(s, p_fw4, wt_fw4, 128, FACEC);

    // ---- normalize direction kernels
    norm_cols3_k<<<(SUP * 128 + 255) / 256, 256, 0, s>>>(p_d0, d0n, SUP * 128);
    norm_cols3_k<<<(SUP * 128 + 255) / 256, 256, 0, s>>>(p_d1, d1n, SUP * 128);
    norm_cols3_k<<<(SUP * 256 + 255) / 256, 256, 0, s>>>(p_d2, d2n, SUP * 256);
    norm_cols3_k<<<(SUP * 256 + 255) / 256, 256, 0, s>>>(p_d3, d3n, SUP * 256);
    norm_cols3_k<<<(SUP * 512 + 255) / 256, 256, 0, s>>>(p_d4, d4n, SUP * 512);

    // ---- level 0: full resolution
    knn_k<<<(unsigned)((M0 + 127) / 128), 128, 0, s>>>(verts, idx0, BSZ, NV, NBRK);
    conv_surface_k<<<(unsigned)M0, 128, 0, s>>>(verts, idx0, d0n, fm0, BSZ, NV, NBRK, 128);
    run_gemm(s, fm0, (int)M0, 128, wt_w1, 1024, p_b1, slab1, Abf);       // fo1
    conv_combine_k<<<(unsigned)M0, 128, 0, s>>>(verts, idx0, d1n, slab1, fm1, BSZ, NV, NBRK, 128);
    run_bn(s, fm1, (int)M0, 128, p_g1, p_be1, mv, 1);
    knn_k<<<(unsigned)((M0 + 127) / 128), 128, 0, s>>>(verts, pidx0, BSZ, NV, 4);
    pool_max4_k<<<(unsigned)((M0 * 128 + 255) / 256), 256, 0, s>>>(fm1, pidx0, pooledF, BSZ, NV, 128);

    // ---- level 1: 512 points
    gather_perm_k<<<(unsigned)((M1 * 3 + 255) / 256), 256, 0, s>>>(verts, perm1, v1b, BSZ, NV, NV1, 3);
    gather_perm_k<<<(unsigned)((M1 * 128 + 255) / 256), 256, 0, s>>>(pooledF, perm1, f1b, BSZ, NV, NV1, 128);
    knn_k<<<(unsigned)((M1 + 127) / 128), 128, 0, s>>>(v1b, idx1, BSZ, NV1, NBRK);
    run_gemm(s, f1b, (int)M1, 128, wt_w2, 2048, p_b2, slab2, Abf);       // fo2
    conv_combine_k<<<(unsigned)M1, 256, 0, s>>>(v1b, idx1, d2n, slab2, fm2, BSZ, NV1, NBRK, 256);
    run_bn(s, fm2, (int)M1, 256, p_g2, p_be2, mv, 1);
    run_gemm(s, fm2, (int)M1, 256, wt_w3, 2048, p_b3, slab3, Abf);       // fo3
    conv_combine_k<<<(unsigned)M1, 256, 0, s>>>(v1b, idx1, d3n, slab3, fm3, BSZ, NV1, NBRK, 256);
    run_bn(s, fm3, (int)M1, 256, p_g3, p_be3, mv, 1);
    knn_k<<<(unsigned)((M1 + 127) / 128), 128, 0, s>>>(v1b, pidx1, BSZ, NV1, 4);
    pool_max4_k<<<(unsigned)((M1 * 256 + 255) / 256), 256, 0, s>>>(fm3, pidx1, pooled1, BSZ, NV1, 256);

    // ---- level 2: 128 points
    gather_perm_k<<<(unsigned)((M2 * 3 + 255) / 256), 256, 0, s>>>(v1b, perm2, v2b, BSZ, NV1, NV2, 3);
    gather_perm_k<<<(unsigned)((M2 * 256 + 255) / 256), 256, 0, s>>>(pooled1, perm2, f2b, BSZ, NV1, NV2, 256);
    knn_k<<<(unsigned)((M2 + 127) / 128), 128, 0, s>>>(v2b, idx2, BSZ, NV2, NBRK);
    run_gemm(s, f2b, (int)M2, 256, wt_w4, 4096, p_b4, slab4, Abf);       // fo4
    conv_combine_k<<<(unsigned)M2, 512, 0, s>>>(v2b, idx2, d4n, slab4, fm4, BSZ, NV2, NBRK, 512);

    // ---- upsample by nearest neighbor
    nearest_idx_k<<<(unsigned)((M0 + 127) / 128), 128, 0, s>>>(verts, v1b, nidx1, BSZ, NV, NV1);
    gather_bv_k<<<(unsigned)((M0 * 256 + 255) / 256), 256, 0, s>>>(fm2, nidx1, fm2u, BSZ, NV, NV1, 256);
    gather_bv_k<<<(unsigned)((M0 * 256 + 255) / 256), 256, 0, s>>>(fm3, nidx1, fm3u, BSZ, NV, NV1, 256);
    nearest_idx_k<<<(unsigned)((M0 + 127) / 128), 128, 0, s>>>(verts, v2b, nidx2, BSZ, NV, NV2);
    gather_bv_k<<<(unsigned)((M0 * 512 + 255) / 256), 256, 0, s>>>(fm4, nidx2, fm4u, BSZ, NV, NV2, 512);

    // ---- feat concat (written directly into d_out feat section, stride 1286)
    copy_seg_k<<<(unsigned)((M0 * 128 + 255) / 256), 256, 0, s>>>(fm0, feat, M0, 128, 1286, 0);
    copy_seg_k<<<(unsigned)((M0 * 128 + 255) / 256), 256, 0, s>>>(fm1, feat, M0, 128, 1286, 128);
    copy_seg_k<<<(unsigned)((M0 * 256 + 255) / 256), 256, 0, s>>>(fm2u, feat, M0, 256, 1286, 256);
    copy_seg_k<<<(unsigned)((M0 * 256 + 255) / 256), 256, 0, s>>>(fm3u, feat, M0, 256, 1286, 512);
    copy_seg_k<<<(unsigned)((M0 * 512 + 255) / 256), 256, 0, s>>>(fm4u, feat, M0, 512, 1286, 768);
    onehot_k<<<(unsigned)((M0 * OBJC + 255) / 256), 256, 0, s>>>(cat_id, feat, BSZ, NV);
    mean_feat_k<<<(BSZ * 1280 + 255) / 256, 256, 0, s>>>(feat, ffb, BSZ, NV);

    // ---- classification/recon head: x chain (slabs reused)
    run_gemm(s, feat, (int)M0, 1286, wt_cw1, 512, p_cb1, slab2, Abf);   // x1
    run_bn(s, slab2, (int)M0, 512, p_cg1, p_ce1, mv, 1);
    run_gemm(s, slab2, (int)M0, 512, wt_cw2, 512, p_cb2, slab3, Abf);   // x2
    run_bn(s, slab3, (int)M0, 512, p_cg2, p_ce2, mv, 1);
    run_gemm(s, slab3, (int)M0, 512, wt_cw3, 256, p_cb3, slab4, Abf);   // x3
    run_bn(s, slab4, (int)M0, 256, p_cg3, p_ce3, mv, 1);
    run_gemm(s, slab4, (int)M0, 256, wt_rw1, 128, p_rb1, pooledF, Abf); // r1
    run_bn(s, pooledF, (int)M0, 128, p_rg1, p_re1, mv, 1);
    run_gemm(s, pooledF, (int)M0, 128, wt_rw2, 3, p_rb2, recon, Abf);   // recon out

    // ---- face head: y chain
    build_y_k<<<(unsigned)((M0 * 1283 + 255) / 256), 256, 0, s>>>(ffb, verts, slab1, BSZ, NV);
    run_gemm(s, slab1, (int)M0, 1283, wt_fw1, 512, p_fb1, fm4u, Abf);   // y1
    run_bn(s, fm4u, (int)M0, 512, p_fg1, p_fe1, mv, 1);
    run_gemm(s, fm4u, (int)M0, 512, wt_fw2, 256, p_fb2, fm2u, Abf);     // y2
    run_bn(s, fm2u, (int)M0, 256, p_fg2, p_fe2, mv, 1);
    run_gemm(s, fm2u, (int)M0, 256, wt_fw3, 128, p_fb3, fm3u, Abf);     // y3
    run_bn(s, fm3u, (int)M0, 128, p_fg3, p_fe3, mv, 1);
    run_gemm(s, fm3u, (int)M0, 128, wt_fw4, FACEC, p_fb4, face, Abf);   // face out
}
